// SerializedKeypointsStackedCANeck_53154515255407
// MI455X (gfx1250) — matrix operand order repaired and checked by reading, not compiled, not run on hardware
//
#include <hip/hip_runtime.h>

#define B_     64
#define C_     128
#define H_     8
#define HD_    16
#define KQ_    16
#define FF_    512
#define DEPTH_ 6
#define NW_    8
#define SPLIT_ 8
#define NPART_ (SPLIT_*NW_)

typedef __attribute__((ext_vector_type(16))) __bf16       v16bf;
typedef __attribute__((ext_vector_type(8)))  float        v8f;
typedef __attribute__((ext_vector_type(4)))  unsigned int u32x4;
typedef __attribute__((ext_vector_type(4)))  float        f32x4;

__device__ __forceinline__ v8f wmma_bf16(v16bf a, v16bf b, v8f c) {
  // D = A(16x32 bf16) x B(32x16 bf16) + C(16x16 f32)
  return __builtin_amdgcn_wmma_f32_16x16x32_bf16(false, a, false, b, (short)0, c, false, false);
}
__device__ __forceinline__ v8f zero8() {
  v8f z = {0.f, 0.f, 0.f, 0.f, 0.f, 0.f, 0.f, 0.f};
  return z;
}
// native exp2: bare v_exp_f32 (no libm denormal fixup, no ln->log2 multiply)
__device__ __forceinline__ float fexp2(float x) { return __builtin_amdgcn_exp2f(x); }

// pack two f32 into a dword of two bf16 (truncating round) with one v_perm_b32
__device__ __forceinline__ unsigned pk_bf16(float lo, float hi) {
  union { float f; unsigned u; } a, b;
  a.f = hi; b.f = lo;
  return __builtin_amdgcn_perm(a.u, b.u, 0x07060302u);
}
// store one f32 as truncated bf16 (ds_store_b16_d16_hi pattern)
__device__ __forceinline__ void st_bf16(__bf16* p, float x) {
  union { float f; unsigned u; } t; t.f = x;
  *(unsigned short*)p = (unsigned short)(t.u >> 16);
}
// load 8 contiguous bf16 (16B-aligned) into elements [o, o+8) of a v16bf
__device__ __forceinline__ void ld8bf(v16bf& d, int o, const __bf16* p) {
  union { u32x4 v; __bf16 h[8]; } t;
  t.v = *(const u32x4*)p;
#pragma unroll
  for (int i = 0; i < 8; ++i) d[o + i] = t.h[i];
}
// load 8 contiguous bf16 into low half, zero high half (dword-level)
__device__ __forceinline__ v16bf ld8bf_pad(const __bf16* p) {
  union { v16bf v; u32x4 q[2]; } t;
  t.q[0] = *(const u32x4*)p;
  t.q[1] = (u32x4){0u, 0u, 0u, 0u};
  return t.v;
}

// ---------------------------------------------------------------- init q state
__global__ __launch_bounds__(256) void k_init(float* qs, const float* ke) {
  int b = blockIdx.x, t = threadIdx.x;
  for (int i = t; i < KQ_ * C_; i += 256) {
    int q = i / C_, c = i % C_;
    qs[b * KQ_ * C_ + i] = (q < 15) ? ke[q * C_ + c] : 0.f;
  }
}

// ------------------------------------------ prepack Wk/Wv into B-fragment bf16
// layout: [m(k=0,v=1)][head j][kchunk kc][lane][e] ; value = W[cin][cout]
__global__ __launch_bounds__(256) void k_prepack(const float* wk, const float* wv, __bf16* wf) {
  int idx = blockIdx.x * 256 + threadIdx.x;
  if (idx >= 2 * H_ * 4 * 32 * 16) return;
  int e    = idx & 15;
  int lane = (idx >> 4) & 31;
  int kc   = (idx >> 9) & 3;
  int j    = (idx >> 11) & 7;
  int m    = (idx >> 14) & 1;
  int cin  = 32 * kc + ((lane < 16) ? 0 : 8) + e + ((e & 8) ? 8 : 0);
  int cout = 16 * j + (lane & 15);
  const float* w = m ? wv : wk;
  wf[idx] = (__bf16)w[cin * C_ + cout];
}

// ------------------------------- q projection (scale = log2(e)/sqrt(hd), bf16)
__global__ __launch_bounds__(256) void k_qproj(const float* qs, const float* wq,
                                               const float* bq, __bf16* qp) {
  int b = blockIdx.x, t = threadIdx.x;
  __shared__ float X[KQ_][C_];
  for (int i = t; i < KQ_ * C_; i += 256) X[i / C_][i % C_] = qs[b * KQ_ * C_ + i];
  __syncthreads();
  for (int i = t; i < KQ_ * C_; i += 256) {
    int q = i / C_, c = i % C_;
    float s = bq[c];
    for (int k = 0; k < C_; ++k) s += X[q][k] * wq[k * C_ + c];
    qp[b * KQ_ * C_ + i] = (__bf16)(0.25f * 1.44269504088896f * s);  // base-2 softmax
  }
}

// ----------------------------------------- self-attn (15 tokens) + residual + LN
__global__ __launch_bounds__(256) void k_sa(float* qs, const float* w, const float* bias,
                                            const float* lnw, const float* lnb) {
  int b = blockIdx.x, t = threadIdx.x;
  __shared__ float X[KQ_][C_], Q[KQ_][C_], Km[KQ_][C_], V[KQ_][C_];
  __shared__ float P[KQ_][H_][KQ_];
  __shared__ float Mu[KQ_], Iv[KQ_];
  for (int i = t; i < KQ_ * C_; i += 256) X[i / C_][i % C_] = qs[b * KQ_ * C_ + i];
  __syncthreads();
  for (int i = t; i < KQ_ * C_; i += 256) {
    int q = i / C_, c = i % C_;
    float aq = bias[0 * C_ + c], ak = bias[1 * C_ + c], av = bias[2 * C_ + c];
    for (int k = 0; k < C_; ++k) {
      float x = X[q][k];
      aq += x * w[0 * C_ * C_ + k * C_ + c];
      ak += x * w[1 * C_ * C_ + k * C_ + c];
      av += x * w[2 * C_ * C_ + k * C_ + c];
    }
    Q[q][c] = aq; Km[q][c] = ak; V[q][c] = av;
  }
  __syncthreads();
  if (t < KQ_ * H_) {
    int q = t / H_, h = t % H_;
    float sc[15]; float mx = -1e30f;
    for (int j = 0; j < 15; ++j) {
      float s = 0.f;
      for (int d2 = 0; d2 < HD_; ++d2) s += Q[q][h * HD_ + d2] * Km[j][h * HD_ + d2];
      s *= 0.25f; sc[j] = s; mx = fmaxf(mx, s);
    }
    float l = 0.f;
    for (int j = 0; j < 15; ++j) { float p = __expf(sc[j] - mx); sc[j] = p; l += p; }
    float inv = 1.f / l;
    for (int j = 0; j < 15; ++j) P[q][h][j] = sc[j] * inv;
  }
  __syncthreads();
  for (int i = t; i < KQ_ * C_; i += 256) {           // attn out -> reuse Km
    int q = i / C_, c = i % C_, h = c / HD_;
    float s = 0.f;
    for (int j = 0; j < 15; ++j) s += P[q][h][j] * V[j][c];
    Km[q][c] = s;
  }
  __syncthreads();
  for (int i = t; i < KQ_ * C_; i += 256) {           // out proj + residual -> Q
    int q = i / C_, c = i % C_;
    float s = bias[3 * C_ + c];
    for (int k = 0; k < C_; ++k) s += Km[q][k] * w[3 * C_ * C_ + k * C_ + c];
    Q[q][c] = s + X[q][c];
  }
  __syncthreads();
  if (t < KQ_) {
    float mu = 0.f; for (int c = 0; c < C_; ++c) mu += Q[t][c]; mu *= (1.f / C_);
    float v2 = 0.f; for (int c = 0; c < C_; ++c) { float d2 = Q[t][c] - mu; v2 += d2 * d2; }
    Mu[t] = mu; Iv[t] = rsqrtf(v2 * (1.f / C_) + 1e-5f);
  }
  __syncthreads();
  for (int i = t; i < KQ_ * C_; i += 256) {
    int q = i / C_, c = i % C_;
    qs[b * KQ_ * C_ + i] = (Q[q][c] - Mu[q]) * Iv[q] * lnw[c] + lnb[c];
  }
}

// ------------------------------------------- fused flash cross-attention (WMMA)
__global__ __launch_bounds__(NW_ * 32) void k_flash(
    const float* __restrict__ m0, const float* __restrict__ m1,
    const float* __restrict__ m2, const float* __restrict__ m3,
    const int* __restrict__ cu0, const int* __restrict__ cu1,
    const int* __restrict__ cu2, const int* __restrict__ cu3,
    const __bf16* __restrict__ wf, const __bf16* __restrict__ qp,
    const float* __restrict__ cab,
    float* __restrict__ pacc, float* __restrict__ pm, float* __restrict__ pl) {
  int b = blockIdx.x / SPLIT_;
  int sp = blockIdx.x % SPLIT_;
  int wave = threadIdx.x >> 5, lane = threadIdx.x & 31;
  int laneq = lane & 15;
  bool lo = lane < 16;

  int o0 = cu0[b], l0 = cu0[b + 1] - o0;
  int o1 = cu1[b], l1 = cu1[b + 1] - o1;
  int o2 = cu2[b], l2 = cu2[b + 1] - o2;
  int o3 = cu3[b], l3 = cu3[b + 1] - o3;
  int len = l0 + l1 + l2 + l3;
  const float* base0 = m0 + (size_t)o0 * C_;
  const float* base1 = m1 + (size_t)o1 * C_;
  const float* base2 = m2 + (size_t)o2 * C_;
  const float* base3 = m3 + (size_t)o3 * C_;

  __shared__ __align__(16) __bf16 lds_k[NW_][32][HD_];   // [tok][c_head]
  __shared__ __align__(16) __bf16 lds_v[NW_][HD_][32];   // [c_head][tok]
  __shared__ __align__(16) __bf16 lds_p[NW_][32][KQ_];   // [tok][q]

  v8f acc[H_]; float mst[H_], lst[H_];
#pragma unroll
  for (int j = 0; j < H_; ++j) {
    mst[j] = -1e30f; lst[j] = 0.f;
#pragma unroll
    for (int v = 0; v < 8; ++v) acc[j][v] = 0.f;
  }

  int rbase = lo ? 0 : 8;
  int ntiles = (len + 31) / 32;
  for (int t = sp * NW_ + wave; t < ntiles; t += SPLIT_ * NW_) {
    int tb = t * 32;
    bool tail = (tb + 32 > len);               // wave-uniform
    // ---- load memory tile as A-fragments (M=tok, K=c_in), fp32->bf16 trunc ----
    v16bf am[2][4];
#pragma unroll
    for (int st = 0; st < 2; ++st) {
      int row = tb + st * 16 + laneq; if (row > len - 1) row = len - 1;
      const float* rp;
      { int p = row;
        if (p < l0) rp = base0 + (size_t)p * C_;
        else { p -= l0; if (p < l1) rp = base1 + (size_t)p * C_;
          else { p -= l1; if (p < l2) rp = base2 + (size_t)p * C_;
            else { p -= l2; rp = base3 + (size_t)p * C_; } } } }
#pragma unroll
      for (int kc = 0; kc < 4; ++kc) {
        int cA = 32 * kc + (lo ? 0 : 8);
        int cB = 32 * kc + (lo ? 16 : 24);
        f32x4 x0 = *(const f32x4*)(rp + cA), x1 = *(const f32x4*)(rp + cA + 4);
        f32x4 x2 = *(const f32x4*)(rp + cB), x3 = *(const f32x4*)(rp + cB + 4);
        union { v16bf v; unsigned u[8]; } F;
        F.u[0] = pk_bf16(x0[0], x0[1]); F.u[1] = pk_bf16(x0[2], x0[3]);
        F.u[2] = pk_bf16(x1[0], x1[1]); F.u[3] = pk_bf16(x1[2], x1[3]);
        F.u[4] = pk_bf16(x2[0], x2[1]); F.u[5] = pk_bf16(x2[2], x2[3]);
        F.u[6] = pk_bf16(x3[0], x3[1]); F.u[7] = pk_bf16(x3[2], x3[3]);
        am[st][kc] = F.v;
      }
    }
#pragma unroll
    for (int j = 0; j < H_; ++j) {
      // ---- K projection (inline-zero C; bias folded at store) ----
      v8f dk0, dk1;
      {
        v16bf wb;
        const __bf16* wp = wf + ((size_t)((0 * H_ + j) * 4 + 0) * 32 + lane) * 16;
        ld8bf(wb, 0, wp); ld8bf(wb, 8, wp + 8);
        dk0 = wmma_bf16(am[0][0], wb, zero8());
        dk1 = wmma_bf16(am[1][0], wb, zero8());
      }
#pragma unroll
      for (int kc = 1; kc < 4; ++kc) {
        v16bf wb;
        const __bf16* wp = wf + ((size_t)((0 * H_ + j) * 4 + kc) * 32 + lane) * 16;
        ld8bf(wb, 0, wp); ld8bf(wb, 8, wp + 8);
        dk0 = wmma_bf16(am[0][kc], wb, dk0);
        dk1 = wmma_bf16(am[1][kc], wb, dk1);
      }
      float bk = cab[C_ + j * HD_ + laneq];
#pragma unroll
      for (int v = 0; v < 8; ++v) {
        st_bf16(&lds_k[wave][rbase + v][laneq],      dk0[v] + bk);
        st_bf16(&lds_k[wave][16 + rbase + v][laneq], dk1[v] + bk);
      }
      asm volatile("s_wait_dscnt 0x0" ::: "memory");
      // ---- scores: s^T = k_tile @ q^T (K padded 16->32 with zeros) ----
      v16bf qb;
      {
        union { v16bf v; u32x4 q[2]; } t2;
        t2.q[0] = (u32x4){0u, 0u, 0u, 0u};
        t2.q[1] = (u32x4){0u, 0u, 0u, 0u};
        qb = t2.v;
      }
      {
        const __bf16* qpp = qp + ((size_t)(b * KQ_ + laneq)) * C_ + j * HD_ + (lo ? 0 : 8);
        ld8bf(qb, 0, qpp);
      }
      v16bf ak0 = ld8bf_pad(&lds_k[wave][laneq][lo ? 0 : 8]);
      v8f s0 = wmma_bf16(ak0, qb, zero8());
      v16bf ak1 = ld8bf_pad(&lds_k[wave][16 + laneq][lo ? 0 : 8]);
      v8f s1 = wmma_bf16(ak1, qb, zero8());
      if (tail) {                                    // scalar-branch masked path
#pragma unroll
        for (int v = 0; v < 8; ++v) {
          int tok0 = tb + rbase + v;
          if (tok0 >= len)      s0[v] = -1e30f;
          if (tok0 + 16 >= len) s1[v] = -1e30f;
        }
      }
      // ---- online softmax, base-2 domain (native v_exp_f32) ----
      float tmax = -1e30f;
#pragma unroll
      for (int v = 0; v < 8; ++v) tmax = fmaxf(tmax, fmaxf(s0[v], s1[v]));
      tmax = fmaxf(tmax, __shfl_xor(tmax, 16));
      float mnew = fmaxf(mst[j], tmax);
      float corr = fexp2(mst[j] - mnew);
      mst[j] = mnew;
      float rsum = 0.f;
#pragma unroll
      for (int v = 0; v < 8; ++v) {
        float p0 = fexp2(s0[v] - mnew), p1 = fexp2(s1[v] - mnew);
        rsum += p0 + p1;
        st_bf16(&lds_p[wave][rbase + v][laneq],      p0);
        st_bf16(&lds_p[wave][16 + rbase + v][laneq], p1);
      }
      rsum += __shfl_xor(rsum, 16);
      lst[j] = lst[j] * corr + rsum;
#pragma unroll
      for (int v = 0; v < 8; ++v) acc[j][v] *= corr;
      // ---- V projection ----
      v8f dv0, dv1;
      {
        v16bf wb;
        const __bf16* wp = wf + ((size_t)((1 * H_ + j) * 4 + 0) * 32 + lane) * 16;
        ld8bf(wb, 0, wp); ld8bf(wb, 8, wp + 8);
        dv0 = wmma_bf16(am[0][0], wb, zero8());
        dv1 = wmma_bf16(am[1][0], wb, zero8());
      }
#pragma unroll
      for (int kc = 1; kc < 4; ++kc) {
        v16bf wb;
        const __bf16* wp = wf + ((size_t)((1 * H_ + j) * 4 + kc) * 32 + lane) * 16;
        ld8bf(wb, 0, wp); ld8bf(wb, 8, wp + 8);
        dv0 = wmma_bf16(am[0][kc], wb, dv0);
        dv1 = wmma_bf16(am[1][kc], wb, dv1);
      }
      float bv = cab[2 * C_ + j * HD_ + laneq];
#pragma unroll
      for (int v = 0; v < 8; ++v) {
        st_bf16(&lds_v[wave][laneq][rbase + v],      dv0[v] + bv);
        st_bf16(&lds_v[wave][laneq][16 + rbase + v], dv1[v] + bv);
      }
      asm volatile("s_wait_dscnt 0x0" ::: "memory");
      // ---- PV: acc^T += v^T @ p (full K=32 over tokens) ----
      v16bf av2, pb;
      ld8bf(av2, 0, &lds_v[wave][laneq][lo ? 0 : 8]);
      ld8bf(av2, 8, &lds_v[wave][laneq][lo ? 16 : 24]);
      {
        int kb = lo ? 0 : 8;
#pragma unroll
        for (int e = 0; e < 16; ++e) pb[e] = lds_p[wave][kb + e + ((e & 8) ? 8 : 0)][laneq];
      }
      acc[j] = wmma_bf16(av2, pb, acc[j]);
    }  // heads
  }    // tiles

  int P = (b * SPLIT_ + sp) * NW_ + wave;
#pragma unroll
  for (int j = 0; j < H_; ++j) {
    float* pa = pacc + (((size_t)(P * H_ + j)) * KQ_ + laneq) * HD_;
#pragma unroll
    for (int v = 0; v < 8; ++v) pa[rbase + v] = acc[j][v];
    if (lo) {
      pm[(P * H_ + j) * KQ_ + laneq] = mst[j];
      pl[(P * H_ + j) * KQ_ + laneq] = lst[j];
    }
  }
}

// ----------------------- merge partials + out-projection + residual + LayerNorm
__global__ __launch_bounds__(256) void k_combine(
    const float* __restrict__ pacc, const float* __restrict__ pm,
    const float* __restrict__ pl, float* __restrict__ qs,
    const float* __restrict__ w, const float* __restrict__ bias,
    const float* __restrict__ lnw, const float* __restrict__ lnb) {
  int b = blockIdx.x, t = threadIdx.x;
  __shared__ float A[KQ_][C_], R[KQ_][C_];
  __shared__ float Ms[H_][KQ_], Ls[H_][KQ_], Mu[KQ_], Iv[KQ_];
  int Pb = b * NPART_;
  if (t < H_ * KQ_) {
    int h = t / KQ_, q = t % KQ_;
    float M = -1e30f;
    for (int p = 0; p < NPART_; ++p) M = fmaxf(M, pm[((Pb + p) * H_ + h) * KQ_ + q]);
    float L = 0.f;
    for (int p = 0; p < NPART_; ++p) {
      int ix = ((Pb + p) * H_ + h) * KQ_ + q;
      L += pl[ix] * fexp2(pm[ix] - M);          // base-2 domain (matches k_flash)
    }
    Ms[h][q] = M; Ls[h][q] = fmaxf(L, 1e-30f);
  }
  __syncthreads();
  for (int i = t; i < KQ_ * C_; i += 256) {
    int q = i / C_, c = i % C_, h = c / HD_, cp = c % HD_;
    float s = 0.f, M = Ms[h][q];
    for (int p = 0; p < NPART_; ++p) {
      int ix = ((Pb + p) * H_ + h) * KQ_ + q;
      s += pacc[(size_t)ix * HD_ + cp] * fexp2(pm[ix] - M);
    }
    A[q][c] = s / Ls[h][q];
  }
  __syncthreads();
  for (int i = t; i < KQ_ * C_; i += 256) {
    int q = i / C_, c = i % C_;
    float s = bias[3 * C_ + c];
    for (int k = 0; k < C_; ++k) s += A[q][k] * w[3 * C_ * C_ + k * C_ + c];
    R[q][c] = s + qs[b * KQ_ * C_ + i];
  }
  __syncthreads();
  if (t < KQ_) {
    float mu = 0.f; for (int c = 0; c < C_; ++c) mu += R[t][c]; mu *= (1.f / C_);
    float v2 = 0.f; for (int c = 0; c < C_; ++c) { float d2 = R[t][c] - mu; v2 += d2 * d2; }
    Mu[t] = mu; Iv[t] = rsqrtf(v2 * (1.f / C_) + 1e-5f);
  }
  __syncthreads();
  for (int i = t; i < KQ_ * C_; i += 256) {
    int q = i / C_, c = i % C_;
    qs[b * KQ_ * C_ + i] = (R[q][c] - Mu[q]) * Iv[q] * lnw[c] + lnb[c];
  }
}

// -------------------------------------------------- FFN + residual + LayerNorm
__global__ __launch_bounds__(256) void k_ffn(float* qs, const float* w1, const float* b1,
                                             const float* w2, const float* b2,
                                             const float* lnw, const float* lnb) {
  int b = blockIdx.x, t = threadIdx.x;
  __shared__ float X[KQ_][C_], Hh[KQ_][FF_], R[KQ_][C_];
  __shared__ float Mu[KQ_], Iv[KQ_];
  for (int i = t; i < KQ_ * C_; i += 256) X[i / C_][i % C_] = qs[b * KQ_ * C_ + i];
  __syncthreads();
  for (int i = t; i < KQ_ * FF_; i += 256) {
    int q = i / FF_, f = i % FF_;
    float s = b1[f];
    for (int k = 0; k < C_; ++k) s += X[q][k] * w1[k * FF_ + f];
    Hh[q][f] = fmaxf(s, 0.f);
  }
  __syncthreads();
  for (int i = t; i < KQ_ * C_; i += 256) {
    int q = i / C_, c = i % C_;
    float s = b2[c];
    for (int f = 0; f < FF_; ++f) s += Hh[q][f] * w2[f * C_ + c];
    R[q][c] = s + X[q][c];
  }
  __syncthreads();
  if (t < KQ_) {
    float mu = 0.f; for (int c = 0; c < C_; ++c) mu += R[t][c]; mu *= (1.f / C_);
    float v2 = 0.f; for (int c = 0; c < C_; ++c) { float d2 = R[t][c] - mu; v2 += d2 * d2; }
    Mu[t] = mu; Iv[t] = rsqrtf(v2 * (1.f / C_) + 1e-5f);
  }
  __syncthreads();
  for (int i = t; i < KQ_ * C_; i += 256) {
    int q = i / C_, c = i % C_;
    qs[b * KQ_ * C_ + i] = (R[q][c] - Mu[q]) * Iv[q] * lnw[c] + lnb[c];
  }
}

// ------------------------------------------------------------ final LN -> d_out
__global__ __launch_bounds__(256) void k_norm(const float* qs, const float* nw,
                                              const float* nb, float* out) {
  int b = blockIdx.x, t = threadIdx.x;
  __shared__ float X[KQ_][C_]; __shared__ float Mu[KQ_], Iv[KQ_];
  for (int i = t; i < KQ_ * C_; i += 256) X[i / C_][i % C_] = qs[b * KQ_ * C_ + i];
  __syncthreads();
  if (t < KQ_) {
    float mu = 0.f; for (int c = 0; c < C_; ++c) mu += X[t][c]; mu *= (1.f / C_);
    float v2 = 0.f; for (int c = 0; c < C_; ++c) { float d2 = X[t][c] - mu; v2 += d2 * d2; }
    Mu[t] = mu; Iv[t] = rsqrtf(v2 * (1.f / C_) + 1e-5f);
  }
  __syncthreads();
  for (int i = t; i < 15 * C_; i += 256) {
    int q = i / C_, c = i % C_;
    out[((size_t)b * 15 + q) * C_ + c] = (X[q][c] - Mu[q]) * Iv[q] * nw[c] + nb[c];
  }
}

// ------------------------------------------------------------------- launcher
extern "C" void kernel_launch(void* const* d_in, const int* in_sizes, int n_in,
                              void* d_out, int out_size, void* d_ws, size_t ws_size,
                              hipStream_t stream) {
  (void)in_sizes; (void)n_in; (void)out_size; (void)ws_size;
  const int*   cu0 = (const int*)d_in[0];
  const float* m0  = (const float*)d_in[1];
  const int*   cu1 = (const int*)d_in[2];
  const float* m1  = (const float*)d_in[3];
  const int*   cu2 = (const int*)d_in[4];
  const float* m2  = (const float*)d_in[5];
  const int*   cu3 = (const int*)d_in[6];
  const float* m3  = (const float*)d_in[7];
  const float* ke  = (const float*)d_in[8];
  const float* saw = (const float*)d_in[9];
  const float* sab = (const float*)d_in[10];
  const float* caw = (const float*)d_in[11];
  const float* cabp= (const float*)d_in[12];
  const float* w1  = (const float*)d_in[13];
  const float* b1  = (const float*)d_in[14];
  const float* w2  = (const float*)d_in[15];
  const float* b2  = (const float*)d_in[16];
  const float* lnw = (const float*)d_in[17];
  const float* lnb = (const float*)d_in[18];
  const float* nw  = (const float*)d_in[19];
  const float* nb  = (const float*)d_in[20];

  char* ws = (char*)d_ws;
  size_t off = 0;
  float*  qs = (float*)(ws + off);  off += (size_t)B_ * KQ_ * C_ * 4;
  __bf16* qp = (__bf16*)(ws + off); off += (size_t)B_ * KQ_ * C_ * 2;
  __bf16* wf = (__bf16*)(ws + off); off += (size_t)2 * H_ * 4 * 32 * 16 * 2;
  off = (off + 255) & ~(size_t)255;
  float* pacc = (float*)(ws + off); off += (size_t)B_ * NPART_ * H_ * KQ_ * HD_ * 4;
  float* pmv  = (float*)(ws + off); off += (size_t)B_ * NPART_ * H_ * KQ_ * 4;
  float* plv  = (float*)(ws + off); off += (size_t)B_ * NPART_ * H_ * KQ_ * 4;

  k_init<<<B_, 256, 0, stream>>>(qs, ke);
  for (int d = 0; d < DEPTH_; ++d) {
    const float* sawd = saw + (size_t)d * 4 * C_ * C_;
    const float* sabd = sab + (size_t)d * 4 * C_;
    const float* cawd = caw + (size_t)d * 4 * C_ * C_;
    const float* cabd = cabp + (size_t)d * 4 * C_;
    const float* l0w = lnw + (size_t)(d * 3 + 0) * C_, *l0b = lnb + (size_t)(d * 3 + 0) * C_;
    const float* l1w = lnw + (size_t)(d * 3 + 1) * C_, *l1b = lnb + (size_t)(d * 3 + 1) * C_;
    const float* l2w = lnw + (size_t)(d * 3 + 2) * C_, *l2b = lnb + (size_t)(d * 3 + 2) * C_;

    k_sa<<<B_, 256, 0, stream>>>(qs, sawd, sabd, l0w, l0b);
    k_prepack<<<(2 * H_ * 4 * 32 * 16) / 256, 256, 0, stream>>>(
        cawd + 1 * C_ * C_, cawd + 2 * C_ * C_, wf);
    k_qproj<<<B_, 256, 0, stream>>>(qs, cawd, cabd, qp);
    k_flash<<<B_ * SPLIT_, NW_ * 32, 0, stream>>>(m0, m1, m2, m3, cu0, cu1, cu2, cu3,
                                                  wf, qp, cabd, pacc, pmv, plv);
    k_combine<<<B_, 256, 0, stream>>>(pacc, pmv, plv, qs, cawd, cabd, l1w, l1b);
    k_ffn<<<B_, 256, 0, stream>>>(qs, w1 + (size_t)d * C_ * FF_, b1 + (size_t)d * FF_,
                                  w2 + (size_t)d * FF_ * C_, b2 + (size_t)d * C_, l2w, l2b);
  }
  k_norm<<<B_, 256, 0, stream>>>(qs, nw, nb, (float*)d_out);
}
